// GPR_ATT_31078383353907
// MI455X (gfx1250) — matrix operand order repaired and checked by reading, not compiled, not run on hardware
//
#include <hip/hip_runtime.h>

// ---------------------------------------------------------------------------
// GPRGNN pipeline for MI455X (gfx1250, wave32).
//   Weights are pre-swizzled ONCE into f16 WMMA B-fragment layout in global
//   memory (64 KB total -> L2/L0 resident), so the GEMMs need no LDS staging
//   and no barriers: each lane reads its 16 B-values as 32 contiguous bytes.
//   K is a template parameter -> fully unrolled, branchless inner loop.
//
//   h      = x @ W_in + b_in                      (WMMA f16->f32)
//   hidden = temp[0] * h
//   repeat 3x:  z = cur @ W_i + b_i               (WMMA)
//               agg = scatter_add(z[src]*w, dst)  (float4 gather + f32 atomics)
//               cur = relu(agg); hidden += temp[i+1]*cur
//   out    = hidden @ W_out + b_out               (WMMA)
// ---------------------------------------------------------------------------

typedef __attribute__((ext_vector_type(16))) _Float16 v16h;
typedef __attribute__((ext_vector_type(8)))  float    v8f;

#define GNN_NODES 50000
#define GNN_EDGES 800000
#define GNN_INCH  256
#define GNN_HID   64

// ---------------------------------------------------------------------------
// Swizzle f32 W[K x 64] -> f16 fragment layout Wf[K/32][4][32][16]:
//   element W[k][n] -> ks = k>>5, j = (k&7)|(((k>>4)&1)<<3),
//                      lane = (n&15)|(((k>>3)&1)<<4), nb = n>>4
// This matches the A-fragment K order (lanes 0-15 hold K=0-7,16-23,
// lanes 16-31 hold K=8-15,24-31, column = lane&15), tiled over nb and ks.
// ---------------------------------------------------------------------------
__global__ void swizzle_w(const float* __restrict__ W,
                          _Float16* __restrict__ Wf, int total /* K*64 */) {
  const int t = blockIdx.x * blockDim.x + threadIdx.x;
  if (t >= total) return;
  const int k  = t >> 6;
  const int n  = t & 63;
  const int ks = k >> 5;
  const int kl = k & 31;
  const int lane = (n & 15) | (((kl >> 3) & 1) << 4);
  const int nb   = n >> 4;
  const int dst  = (((ks * 4 + nb) * 32) + lane) * 16 + ((kl & 7) | ((kl >> 4) << 3));
  Wf[dst] = (_Float16)W[t];
}

// ---------------------------------------------------------------------------
// C[M x 64] = A[M x K] @ W[K x 64] + bias.  Optionally C2 = temp[tidx] * C.
// Wf is the pre-swizzled f16 weight.  One wave computes a 16x64 tile via
// 4x v_wmma_f32_16x16x32_f16 per k-step; no LDS, no barriers, no branches
// in the K loop (load rows clamped to M-1; stores guarded).
// ---------------------------------------------------------------------------
template <int K>
__global__ __launch_bounds__(256) void gemm64_wmma(
    const float* __restrict__ A, const _Float16* __restrict__ Wf,
    const float* __restrict__ bias, float* __restrict__ C,
    float* __restrict__ C2, const float* __restrict__ temp, int tidx, int M)
{
  const int lane   = threadIdx.x & 31;
  const int wave   = threadIdx.x >> 5;
  const int tile_m = (blockIdx.x * 8 + wave) * 16;

  // A-fragment lane mapping (16-bit A 16x32, ISA 7.12.2):
  //   lanes 0-15: M = lane,    K elems {kh..kh+7, 16+kh..16+kh+7}, kh = 0
  //   lanes16-31: M = lane-16, same with kh = 8
  const int m   = lane & 15;
  const int kh  = (lane >> 4) * 8;
  const int row = tile_m + m;
  // Clamp: every lane loads a valid row; padded rows never stored.
  const int lrow = (row < M) ? row : (M - 1);

  v8f acc[4] = {v8f{}, v8f{}, v8f{}, v8f{}};

  const float*    ap0 = A + (size_t)lrow * K + kh;
  const _Float16* wp0 = Wf + (size_t)lane * 16;

  #pragma unroll
  for (int ks = 0; ks < K / 32; ++ks) {
    // Issue all four B fragments first (8x global_load_b128, one clause;
    // WMMAs below can start on partial loadcnt waits).
    const _Float16* wp = wp0 + ks * (4 * 512);
    const v16h bf0 = *reinterpret_cast<const v16h*>(wp);
    const v16h bf1 = *reinterpret_cast<const v16h*>(wp + 512);
    const v16h bf2 = *reinterpret_cast<const v16h*>(wp + 1024);
    const v16h bf3 = *reinterpret_cast<const v16h*>(wp + 1536);

    // A fragment: 16 f16 values per lane from 4 coalesced float4 loads.
    const float* ap = ap0 + ks * 32;
    const float4 a0 = *reinterpret_cast<const float4*>(ap);
    const float4 a1 = *reinterpret_cast<const float4*>(ap + 4);
    const float4 a2 = *reinterpret_cast<const float4*>(ap + 16);
    const float4 a3 = *reinterpret_cast<const float4*>(ap + 20);
    v16h af;
    af[0]  = (_Float16)a0.x; af[1]  = (_Float16)a0.y;
    af[2]  = (_Float16)a0.z; af[3]  = (_Float16)a0.w;
    af[4]  = (_Float16)a1.x; af[5]  = (_Float16)a1.y;
    af[6]  = (_Float16)a1.z; af[7]  = (_Float16)a1.w;
    af[8]  = (_Float16)a2.x; af[9]  = (_Float16)a2.y;
    af[10] = (_Float16)a2.z; af[11] = (_Float16)a2.w;
    af[12] = (_Float16)a3.x; af[13] = (_Float16)a3.y;
    af[14] = (_Float16)a3.z; af[15] = (_Float16)a3.w;

    acc[0] = __builtin_amdgcn_wmma_f32_16x16x32_f16(
        false, af, false, bf0, (short)0, acc[0], false, false);
    acc[1] = __builtin_amdgcn_wmma_f32_16x16x32_f16(
        false, af, false, bf1, (short)0, acc[1], false, false);
    acc[2] = __builtin_amdgcn_wmma_f32_16x16x32_f16(
        false, af, false, bf2, (short)0, acc[2], false, false);
    acc[3] = __builtin_amdgcn_wmma_f32_16x16x32_f16(
        false, af, false, bf3, (short)0, acc[3], false, false);
  }

  // Epilogue per ISA C/D layout: VGPR r -> M = r + 8*(lane>=16), N = lane&15.
  const int mh = (lane >> 4) * 8;
  const float s2 = (C2 != nullptr) ? temp[tidx] : 0.0f;
  #pragma unroll
  for (int nb = 0; nb < 4; ++nb) {
    const int col = nb * 16 + (lane & 15);
    const float bv = bias[col];
    #pragma unroll
    for (int r = 0; r < 8; ++r) {
      const int orow = tile_m + mh + r;
      if (orow < M) {
        const float v = acc[nb][r] + bv;
        C[(size_t)orow * 64 + col] = v;
        if (C2) C2[(size_t)orow * 64 + col] = s2 * v;
      }
    }
  }
}

// ---------------------------------------------------------------------------
// agg = 0   (float4 stores)
// ---------------------------------------------------------------------------
__global__ void zero_f32v4(float4* __restrict__ p, int n4) {
  int i = blockIdx.x * blockDim.x + threadIdx.x;
  if (i < n4) p[i] = float4{0.0f, 0.0f, 0.0f, 0.0f};
}

// ---------------------------------------------------------------------------
// Edge scatter: agg[dst] += z[src] * w.  16 threads per edge, float4 gather,
// 4x global_atomic_add_f32 per thread (L2 atomics; agg is L2-resident).
// ---------------------------------------------------------------------------
__global__ __launch_bounds__(256) void scatter_edges(
    const int* __restrict__ ei, const float* __restrict__ ew,
    const float* __restrict__ z, float* __restrict__ agg, int nE)
{
  const int t = blockIdx.x * 256 + threadIdx.x;
  const int e = t >> 4;
  if (e >= nE) return;
  const int fq = (t & 15) * 4;

  // Warm future edge metadata into cache (global_prefetch_b8 path).
  __builtin_prefetch(ei + e + 4096, 0, 0);

  const int   s = ei[e];
  const int   d = ei[nE + e];
  const float w = ew[e];

  const float4 v = *reinterpret_cast<const float4*>(z + (size_t)s * 64 + fq);
  float* out = agg + (size_t)d * 64 + fq;
  atomicAdd(out + 0, v.x * w);
  atomicAdd(out + 1, v.y * w);
  atomicAdd(out + 2, v.z * w);
  atomicAdd(out + 3, v.w * w);
}

// ---------------------------------------------------------------------------
// cur = relu(agg); hidden += temp[tidx] * cur   (float4)
// ---------------------------------------------------------------------------
__global__ void relu_combine(const float4* __restrict__ agg,
                             float4* __restrict__ cur,
                             float4* __restrict__ hidden,
                             const float* __restrict__ temp, int tidx, int n4)
{
  int i = blockIdx.x * blockDim.x + threadIdx.x;
  if (i < n4) {
    const float  tv = temp[tidx];
    const float4 a  = agg[i];
    const float4 c  = float4{fmaxf(a.x, 0.0f), fmaxf(a.y, 0.0f),
                             fmaxf(a.z, 0.0f), fmaxf(a.w, 0.0f)};
    cur[i] = c;
    float4 h = hidden[i];
    h.x += tv * c.x; h.y += tv * c.y; h.z += tv * c.z; h.w += tv * c.w;
    hidden[i] = h;
  }
}

// ---------------------------------------------------------------------------
extern "C" void kernel_launch(void* const* d_in, const int* in_sizes, int n_in,
                              void* d_out, int out_size, void* d_ws, size_t ws_size,
                              hipStream_t stream) {
  const float* x       = (const float*)d_in[0];
  const int*   ei      = (const int*)  d_in[1];
  const float* ew      = (const float*)d_in[2];
  const float* W_in    = (const float*)d_in[3];
  const float* b_in    = (const float*)d_in[4];
  const float* layer_W = (const float*)d_in[5];
  const float* layer_b = (const float*)d_in[6];
  const float* W_out   = (const float*)d_in[7];
  const float* b_out   = (const float*)d_in[8];
  const float* temp    = (const float*)d_in[9];
  float* out = (float*)d_out;

  const int M  = GNN_NODES;
  const int NH = GNN_NODES * GNN_HID;   // 3.2M floats per buffer
  const int N4 = NH / 4;

  float* cur    = (float*)d_ws;         // [N, 64]
  float* z      = cur    + NH;          // [N, 64]
  float* agg    = z      + NH;          // [N, 64]
  float* hidden = agg    + NH;          // [N, 64]

  // f16 fragment-layout weights after the f32 buffers (~64 KB)
  _Float16* wf_in  = (_Float16*)(hidden + NH);          // 256*64
  _Float16* wf_l   = wf_in + GNN_INCH * GNN_HID;        // 3 * 64*64
  _Float16* wf_out = wf_l + 3 * GNN_HID * GNN_HID;      // 64*64

  const dim3 gemm_grid((M + 127) / 128);
  const dim3 blk256(256);
  const dim3 ew_grid((N4 + 255) / 256);
  const dim3 sc_grid((GNN_EDGES * 16 + 255) / 256);

  // One-shot weight swizzles (f32 -> f16 fragment layout).
  swizzle_w<<<(GNN_INCH * GNN_HID + 255) / 256, blk256, 0, stream>>>(
      W_in, wf_in, GNN_INCH * GNN_HID);
  for (int i = 0; i < 3; ++i) {
    swizzle_w<<<(GNN_HID * GNN_HID + 255) / 256, blk256, 0, stream>>>(
        layer_W + (size_t)i * GNN_HID * GNN_HID,
        wf_l + (size_t)i * GNN_HID * GNN_HID, GNN_HID * GNN_HID);
  }
  swizzle_w<<<(GNN_HID * GNN_HID + 255) / 256, blk256, 0, stream>>>(
      W_out, wf_out, GNN_HID * GNN_HID);

  // h = x @ W_in + b_in ; hidden = temp[0] * h
  gemm64_wmma<GNN_INCH><<<gemm_grid, blk256, 0, stream>>>(
      x, wf_in, b_in, cur, hidden, temp, 0, M);

  for (int i = 0; i < 3; ++i) {
    // z = cur @ layer_W[i] + layer_b[i]
    gemm64_wmma<GNN_HID><<<gemm_grid, blk256, 0, stream>>>(
        cur, wf_l + (size_t)i * GNN_HID * GNN_HID,
        layer_b + (size_t)i * GNN_HID, z, nullptr, temp, 0, M);
    // agg = 0
    zero_f32v4<<<ew_grid, blk256, 0, stream>>>((float4*)agg, N4);
    // agg[dst] += z[src] * w
    scatter_edges<<<sc_grid, blk256, 0, stream>>>(ei, ew, z, agg, GNN_EDGES);
    // cur = relu(agg); hidden += temp[i+1] * cur
    relu_combine<<<ew_grid, blk256, 0, stream>>>(
        (const float4*)agg, (float4*)cur, (float4*)hidden, temp, i + 1, N4);
  }

  // out = hidden @ W_out + b_out
  gemm64_wmma<GNN_HID><<<gemm_grid, blk256, 0, stream>>>(
      hidden, wf_out, b_out, out, nullptr, temp, 0, M);
}
